// GraphConvolution_8684423872665
// MI455X (gfx1250) — matrix operand order repaired and checked by reading, not compiled, not run on hardware
//
#include <hip/hip_runtime.h>

// ---------------------------------------------------------------------------
// GCN 3-layer forward for MI455X (gfx1250, wave32, WMMA).
//   h = sigmoid( spmm(A, x) @ W1^T + b1 )
//   h = sigmoid( spmm(A, h) @ W2^T + b2 )
//   h =          spmm(A, h) @ W3^T + b3
//   out = softmax(h, axis=1)
// D = 128 features. GEMM uses v_wmma_f32_16x16x32_f16 with LDS-staged tiles.
// ---------------------------------------------------------------------------

#define FD 128            // feature dim
#define LDSS 136          // padded LDS row stride in halves (272B -> bank-conflict-free)

typedef _Float16 v16h __attribute__((ext_vector_type(16)));
typedef _Float16 v8h  __attribute__((ext_vector_type(8)));
typedef float    v8f  __attribute__((ext_vector_type(8)));
typedef __fp16   p2h  __attribute__((ext_vector_type(2)));

// Pack 4 floats -> 4 halves (2x v_cvt_pk_rtz_f16_f32) and store as one b64.
__device__ __forceinline__ void cvt_store4(_Float16* p, float4 v) {
  p2h a = __builtin_amdgcn_cvt_pkrtz(v.x, v.y);
  p2h b = __builtin_amdgcn_cvt_pkrtz(v.z, v.w);
  uint2 u = make_uint2(__builtin_bit_cast(unsigned, a),
                       __builtin_bit_cast(unsigned, b));
  *(uint2*)p = u;
}

// Fast sigmoid: v_exp_f32 + v_rcp_f32 (avoid IEEE div_scale/div_fixup chain).
__device__ __forceinline__ float fast_sigmoid(float x) {
  return __builtin_amdgcn_rcpf(1.0f + __expf(-x));
}

// ---------------------------------------------------------------- zero fill
__global__ __launch_bounds__(256)
void zero_kernel(float* __restrict__ p, long n4) {
  long i = (long)blockIdx.x * blockDim.x + threadIdx.x;
  long stride = (long)gridDim.x * blockDim.x;
  float4 z = make_float4(0.f, 0.f, 0.f, 0.f);
  for (; i < n4; i += stride) ((float4*)p)[i] = z;
}

// --------------------------------------------------------------------- SpMM
// out[dst[e], :] += val[e] * h[src[e], :]   (one edge per wave32,
// 32 lanes x float4 = 128 features, coalesced row load + f32 atomic scatter)
__global__ __launch_bounds__(256)
void spmm_kernel(const float* __restrict__ h,
                 const float* __restrict__ val,
                 const int*   __restrict__ src,
                 const int*   __restrict__ dst,
                 float*       __restrict__ out, int E) {
  const int lane = threadIdx.x & 31;
  int wave = blockIdx.x * (blockDim.x >> 5) + (threadIdx.x >> 5);
  const int nw = gridDim.x * (blockDim.x >> 5);
  for (int e = wave; e < E; e += nw) {
    const int   s = src[e];
    const int   d = dst[e];
    const float v = val[e];
    float4 hv = ((const float4*)(h + (size_t)s * FD))[lane];
    float* o = out + (size_t)d * FD + lane * 4;
    atomicAdd(o + 0, hv.x * v);
    atomicAdd(o + 1, hv.y * v);
    atomicAdd(o + 2, hv.z * v);
    atomicAdd(o + 3, hv.w * v);
  }
}

// ------------------------------------------------- GEMM + bias (+ sigmoid)
// Out = act(A @ W^T + b).  A:[N][128] f32, W:[128][128] f32 row-major.
// Block = 256 thr (8 waves) -> 128 output rows.  A-block and W staged in LDS
// as f16; each wave: 16-row strip, 8 j-tiles x 4 k-chunks of
// v_wmma_f32_16x16x32_f16.  Safe to run with Out == A (full A block is in
// LDS before any store).
template <bool SIGMOID>
__global__ __launch_bounds__(256)
void gemm_bias_act(const float* __restrict__ A,
                   const float* __restrict__ W,
                   const float* __restrict__ bias,
                   float*       __restrict__ Out, int N) {
  __shared__ _Float16 Ah[128 * LDSS];
  __shared__ _Float16 Wh[128 * LDSS];
  const int tid  = threadIdx.x;
  const int row0 = blockIdx.x * 128;
  const int m    = tid & 15;

  // Per-lane bias values for the 8 j-tiles (batched, overlaps staging).
  float bcol[8];
#pragma unroll
  for (int jt = 0; jt < 8; ++jt) bcol[jt] = bias[jt * 16 + m];

  // Stage W (128x128 f32 -> f16 in LDS), coalesced float4 reads.
  for (int idx = tid; idx < 128 * 32; idx += 256) {
    const int r = idx >> 5, c4 = idx & 31;
    float4 w4 = ((const float4*)(W + r * FD))[c4];
    cvt_store4(&Wh[r * LDSS + c4 * 4], w4);
  }
  // Stage this block's 128 rows of A (zero-pad past N).
  for (int idx = tid; idx < 128 * 32; idx += 256) {
    const int r = idx >> 5, c4 = idx & 31;
    const int gr = row0 + r;
    float4 a4 = (gr < N) ? ((const float4*)(A + (size_t)gr * FD))[c4]
                         : make_float4(0.f, 0.f, 0.f, 0.f);
    cvt_store4(&Ah[r * LDSS + c4 * 4], a4);
  }
  __syncthreads();

  const int lane = tid & 31;
  const int wv   = tid >> 5;       // wave id -> 16-row strip
  const int hi   = lane >> 4;      // lane half (0/1)

  // A fragments for this wave's strip, all 4 K-chunks (ISA 16-bit A layout:
  // lane m = M, elems 0..7 -> K=kb..kb+7, elems 8..15 -> K=kb+16..kb+23,
  // kb = kc*32 + 8*hi).
  const _Float16* arow = &Ah[(wv * 16 + m) * LDSS];
  v16h afrag[4];
#pragma unroll
  for (int kc = 0; kc < 4; ++kc) {
    const int kb = kc * 32 + hi * 8;
    v8h lo  = *(const v8h*)(arow + kb);
    v8h hi8 = *(const v8h*)(arow + kb + 16);
    afrag[kc] = __builtin_shufflevector(lo, hi8,
        0, 1, 2, 3, 4, 5, 6, 7, 8, 9, 10, 11, 12, 13, 14, 15);
  }

  const bool full = (row0 + 128 <= N);   // block-uniform: no store guards
  const int  rbase = row0 + wv * 16 + hi * 8;

#pragma unroll
  for (int jt = 0; jt < 8; ++jt) {
    v8f acc = {};
#pragma unroll
    for (int kc = 0; kc < 4; ++kc) {
      // B[k][j] = W[j][k]: lane n = N index (row j of W), elems 0..7 -> K=kb..kb+7,
      // elems 8..15 -> K=kb+16..kb+23, kb = kc*32 + 8*hi (same order as A).
      const _Float16* brow = &Wh[(jt * 16 + m) * LDSS + kc * 32 + hi * 8];
      v8h blo = *(const v8h*)(brow);
      v8h bhi = *(const v8h*)(brow + 16);
      v16h bfrag = __builtin_shufflevector(blo, bhi,
          0, 1, 2, 3, 4, 5, 6, 7, 8, 9, 10, 11, 12, 13, 14, 15);
      acc = __builtin_amdgcn_wmma_f32_16x16x32_f16(
          /*neg_a=*/false, afrag[kc], /*neg_b=*/false, bfrag,
          /*c_mod=*/(short)0, acc, /*reuse_a=*/false, /*reuse_b=*/false);
    }
    // C/D layout: VGPR r -> M = rbase + r, N = lane & 15.
    const int   col = jt * 16 + m;
    const float bb  = bcol[jt];
    float res[8];
#pragma unroll
    for (int r = 0; r < 8; ++r) {
      float v = acc[r] + bb;
      if (SIGMOID) v = fast_sigmoid(v);
      res[r] = v;
    }
    float* obase = Out + (size_t)rbase * FD + col;
    if (full) {
#pragma unroll
      for (int r = 0; r < 8; ++r) obase[r * FD] = res[r];   // imm offsets
    } else {
#pragma unroll
      for (int r = 0; r < 8; ++r)
        if (rbase + r < N) obase[r * FD] = res[r];
    }
  }
}

// ---------------------------------------------------------- row softmax
// One wave per row; 32 lanes x float4 = 128 features; shfl_xor reductions.
__global__ __launch_bounds__(256)
void softmax_kernel(const float* __restrict__ in, float* __restrict__ out, int N) {
  const int row = blockIdx.x * (blockDim.x >> 5) + (threadIdx.x >> 5);
  if (row >= N) return;
  const int lane = threadIdx.x & 31;
  float4 x = ((const float4*)(in + (size_t)row * FD))[lane];
  float mx = fmaxf(fmaxf(x.x, x.y), fmaxf(x.z, x.w));
#pragma unroll
  for (int off = 16; off > 0; off >>= 1)
    mx = fmaxf(mx, __shfl_xor(mx, off, 32));
  float e0 = __expf(x.x - mx), e1 = __expf(x.y - mx);
  float e2 = __expf(x.z - mx), e3 = __expf(x.w - mx);
  float s = e0 + e1 + e2 + e3;
#pragma unroll
  for (int off = 16; off > 0; off >>= 1)
    s += __shfl_xor(s, off, 32);
  const float inv = __builtin_amdgcn_rcpf(s);
  ((float4*)(out + (size_t)row * FD))[lane] =
      make_float4(e0 * inv, e1 * inv, e2 * inv, e3 * inv);
}

// ---------------------------------------------------------------------------
extern "C" void kernel_launch(void* const* d_in, const int* in_sizes, int n_in,
                              void* d_out, int out_size, void* d_ws, size_t ws_size,
                              hipStream_t stream) {
  const float* x   = (const float*)d_in[0];
  const float* ev  = (const float*)d_in[1];
  const float* W1  = (const float*)d_in[2];
  const float* b1  = (const float*)d_in[3];
  const float* W2  = (const float*)d_in[4];
  const float* b2  = (const float*)d_in[5];
  const float* W3  = (const float*)d_in[6];
  const float* b3  = (const float*)d_in[7];
  const int*   src = (const int*)d_in[8];
  const int*   dst = (const int*)d_in[9];

  const int N = in_sizes[0] / FD;
  const int E = in_sizes[1];

  float* outp = (float*)d_out;
  float* tmp  = (float*)d_ws;          // N*FD floats of scratch

  const long n4  = (long)N * FD / 4;
  const int  zb  = 2048;               // zero-fill blocks
  const int  spb = 2048;               // spmm blocks (16384 waves, grid-stride)
  const int  gb  = (N + 127) / 128;    // gemm blocks
  const int  smb = (N + 7) / 8;        // softmax blocks (8 rows/block)

  // ---- layer 1: tmp = sigmoid(spmm(x) @ W1^T + b1)   (gemm in-place on tmp)
  zero_kernel<<<zb, 256, 0, stream>>>(tmp, n4);
  spmm_kernel<<<spb, 256, 0, stream>>>(x, ev, src, dst, tmp, E);
  gemm_bias_act<true><<<gb, 256, 0, stream>>>(tmp, W1, b1, tmp, N);

  // ---- layer 2: out = sigmoid(spmm(tmp) @ W2^T + b2) (gemm in-place on out)
  zero_kernel<<<zb, 256, 0, stream>>>(outp, n4);
  spmm_kernel<<<spb, 256, 0, stream>>>(tmp, ev, src, dst, outp, E);
  gemm_bias_act<true><<<gb, 256, 0, stream>>>(outp, W2, b2, outp, N);

  // ---- layer 3: tmp = spmm(out) @ W3^T + b3; out = softmax(tmp)
  zero_kernel<<<zb, 256, 0, stream>>>(tmp, n4);
  spmm_kernel<<<spb, 256, 0, stream>>>(outp, ev, src, dst, tmp, E);
  gemm_bias_act<false><<<gb, 256, 0, stream>>>(tmp, W3, b3, tmp, N);
  softmax_kernel<<<smb, 256, 0, stream>>>(tmp, outp, N);
}
